// SimpleDecoderLayer_10393820857175
// MI455X (gfx1250) — matrix operand order repaired and checked by reading, not compiled, not run on hardware
//
#include <hip/hip_runtime.h>
#include <hip/hip_bf16.h>

typedef unsigned short u16;
typedef unsigned int   u32;

typedef __attribute__((ext_vector_type(16))) __bf16 v16bf;
typedef __attribute__((ext_vector_type(8)))  float  v8f;
typedef __attribute__((ext_vector_type(4)))  int    v4i;

// Problem dims (fixed by the reference)
constexpr int Bn = 4, Sn = 1024, Dn = 1024, Hn = 16, HDn = 64, FFn = 4096;
constexpr int Mn = Bn * Sn;   // 4096 rows

__device__ __forceinline__ u16 f2bf(float f) {
  u32 u = __float_as_uint(f);
  u += 0x7FFFu + ((u >> 16) & 1u);   // round-to-nearest-even
  return (u16)(u >> 16);
}

union FragU {
  v16bf v;
  u16   u[16];
  uint4 q[2];
};

// ---------------------------------------------------------------------------
// CDNA5 async global->LDS copy (ASYNCcnt path), 16 bytes per lane.
// Builtin signature (from hipcc diagnostics): first arg is int4* in the
// device (AS1) address space; LDS side is AS3. Generic pointers are converted
// via integers: a generic global VA equals the AS1 VA, and the low 32 bits of
// a generic LDS VA are the LDS byte offset (ISA aperture rules).
// ---------------------------------------------------------------------------
typedef __attribute__((address_space(1))) v4i as1_v4i;
typedef __attribute__((address_space(3))) v4i as3_v4i;

#if defined(__has_builtin)
#if __has_builtin(__builtin_amdgcn_global_load_async_to_lds_b128)
#define HAVE_ASYNC_LDS 1
#endif
#endif

__device__ __forceinline__ void async_copy16(const u16* g, u16* l) {
#ifdef HAVE_ASYNC_LDS
  as1_v4i* gp = (as1_v4i*)(size_t)g;
  as3_v4i* lp = (as3_v4i*)(u32)(size_t)l;
  __builtin_amdgcn_global_load_async_to_lds_b128(gp, lp, 0, 0);
#else
  *(uint4*)l = *(const uint4*)g;
#endif
}

__device__ __forceinline__ void wait_async_le8() {
  asm volatile("s_wait_asynccnt 0x8" ::: "memory");
}
__device__ __forceinline__ void wait_async_0() {
  asm volatile("s_wait_asynccnt 0x0" ::: "memory");
}

// ---------------------------------------------------------------------------
// fp32 -> bf16 conversion
// ---------------------------------------------------------------------------
__global__ __launch_bounds__(256) void cvt_kernel(const float* __restrict__ in,
                                                  u16* __restrict__ out, int n) {
  int i = blockIdx.x * blockDim.x + threadIdx.x;
  int stride = gridDim.x * blockDim.x;
  for (; i < n; i += stride) out[i] = f2bf(in[i]);
}

// ---------------------------------------------------------------------------
// Tiled bf16 GEMM:  out[M,N] = A[M,K] @ W[N,K]^T + bias  (both row-major, NT)
// Block tile 128x128, 4 waves (wave32), wave tile 64x64 (16 WMMA acc tiles).
// K staged in double-buffered LDS slabs of 32 via async global->LDS copies.
// MODE 0: fp32 out.  MODE 1: bf16 out in [B,H,S,HD] head layout (Q/K/V).
// MODE 2: exact-GELU -> bf16 out.
// ---------------------------------------------------------------------------
enum { MODE_F32 = 0, MODE_HEAD = 1, MODE_GELU = 2 };

template <int M, int N, int K, int MODE>
__global__ __launch_bounds__(128, 1)
void gemm_kernel(const u16* __restrict__ A, const u16* __restrict__ W,
                 const float* __restrict__ bias, float* __restrict__ outf,
                 u16* __restrict__ outb, float scale) {
  constexpr int KT = 32, LD = KT + 8;  // +8 bf16 pad vs LDS bank conflicts
  __shared__ __align__(16) u16 sA[2][128 * LD];
  __shared__ __align__(16) u16 sB[2][128 * LD];

  const int tid  = threadIdx.x;
  const int lane = tid & 31, wave = tid >> 5;
  const int g = lane >> 4, lr = lane & 15;       // lane half / lane row
  const int m0 = blockIdx.y * 128, n0 = blockIdx.x * 128;
  const int wm = (wave >> 1) * 64, wn = (wave & 1) * 64;

  // per-thread staging pointers/offsets (4 x 16B chunks each for A and W)
  const u16* pa[4];
  const u16* pb[4];
  int lo[4];
#pragma unroll
  for (int i = 0; i < 4; i++) {
    int idx = tid + i * 128;                 // 0..511
    int row = idx >> 2, cc = (idx & 3) * 8;  // 4 x 16B chunks per row
    pa[i] = A + (size_t)(m0 + row) * K + cc;
    pb[i] = W + (size_t)(n0 + row) * K + cc;
    lo[i] = row * LD + cc;
  }
  auto stage = [&](int b) {
#pragma unroll
    for (int i = 0; i < 4; i++) {
      async_copy16(pa[i], &sA[b][lo[i]]);
      async_copy16(pb[i], &sB[b][lo[i]]);
      __builtin_prefetch(pa[i] + KT, 0, 1);  // speculative L2 prefetch, next slab
      __builtin_prefetch(pb[i] + KT, 0, 1);
      pa[i] += KT;
      pb[i] += KT;
    }
  };

  v8f acc[4][4];
#pragma unroll
  for (int i = 0; i < 4; i++)
#pragma unroll
    for (int j = 0; j < 4; j++)
#pragma unroll
      for (int e = 0; e < 8; e++) acc[i][j][e] = 0.0f;

  stage(0);
  int buf = 0;
  for (int k0 = 0; k0 < K; k0 += KT) {
    const bool has_next = (k0 + KT < K);
    if (has_next) stage(buf ^ 1);            // overlap next slab with compute
    if (has_next) wait_async_le8();          // drain current slab (8 async ops)
    else          wait_async_0();
    __syncthreads();

    // Fragments per the ISA 16-bit layouts:
    //  A lane(l): row M = l%16, ks = [8g..8g+7] U [8g+16..8g+23]
    //  B lane(l): col N = l%16, ks = [8g..8g+7] U [8g+16..8g+23]
    FragU a[4], b[4];
#pragma unroll
    for (int mt = 0; mt < 4; mt++) {
      int r = wm + mt * 16 + lr;
      a[mt].q[0] = *(const uint4*)&sA[buf][r * LD + g * 8];
      a[mt].q[1] = *(const uint4*)&sA[buf][r * LD + g * 8 + 16];
    }
#pragma unroll
    for (int nt = 0; nt < 4; nt++) {
      int r = wn + nt * 16 + lr;
      b[nt].q[0] = *(const uint4*)&sB[buf][r * LD + g * 8];
      b[nt].q[1] = *(const uint4*)&sB[buf][r * LD + g * 8 + 16];
    }
#pragma unroll
    for (int mt = 0; mt < 4; mt++)
#pragma unroll
      for (int nt = 0; nt < 4; nt++)
        acc[mt][nt] = __builtin_amdgcn_wmma_f32_16x16x32_bf16(
            false, a[mt].v, false, b[nt].v, (short)0, acc[mt][nt], false, false);
    __syncthreads();   // everyone done reading buf before it is overwritten
    buf ^= 1;
  }

  // epilogue: C layout -> lane holds col n=..+lr, rows M = r + 8g
#pragma unroll
  for (int mt = 0; mt < 4; mt++) {
#pragma unroll
    for (int nt = 0; nt < 4; nt++) {
      int n = n0 + wn + nt * 16 + lr;
      float bv = bias[n];
#pragma unroll
      for (int r = 0; r < 8; r++) {
        int m = m0 + wm + mt * 16 + g * 8 + r;
        float v = (acc[mt][nt][r] + bv) * scale;
        if (MODE == MODE_F32) {
          outf[(size_t)m * N + n] = v;
        } else if (MODE == MODE_HEAD) {
          int h = n >> 6, d = n & 63;            // head / head-dim
          int bb = m >> 10, s = m & 1023;        // batch / seq
          outb[(((size_t)(bb * Hn + h) * Sn + s) << 6) + d] = f2bf(v);
        } else {  // exact GELU (matches jax approximate=False)
          float ge = 0.5f * v * (1.0f + erff(v * 0.70710678118654752f));
          outb[(size_t)m * N + n] = f2bf(ge);
        }
      }
    }
  }
}

// ---------------------------------------------------------------------------
// Causal flash attention. Q/K/V bf16 in [B,H,S,HD] layout, Q pre-scaled by
// 1/sqrt(HD). One 16-query tile per wave; 4 waves/WG share K/V chunks (32
// keys) staged in LDS (V transposed on store). Scores computed transposed
// (S^T = K @ Q^T) so online softmax is lane-local, and the P fragment for
// P@V is a pure lane-local repack of the score accumulators.
// ---------------------------------------------------------------------------
__global__ __launch_bounds__(128, 1)
void attn_kernel(const u16* __restrict__ Qh, const u16* __restrict__ Kh,
                 const u16* __restrict__ Vh, u16* __restrict__ Ctx) {
  constexpr int LDK = HDn + 8;  // 72
  constexpr int LDV = 32 + 8;   // 40
  __shared__ __align__(16) u16 sK[32 * LDK];
  __shared__ __align__(16) u16 sVt[HDn * LDV];

  const int tid = threadIdx.x, lane = tid & 31, wave = tid >> 5;
  const int g = lane >> 4, lr = lane & 15;
  const int bh = blockIdx.y;                 // b*H + h
  const int qblk = blockIdx.x * 64;
  const int q0 = qblk + wave * 16;
  const int q = q0 + lr;                     // this lane's query (C-layout col)

  // Q as B-fragments for S^T = K @ Q^T (two k-steps over HD=64)
  const u16* Qrow = Qh + ((size_t)bh * Sn + q) * HDn;
  FragU bq[2];
#pragma unroll
  for (int f = 0; f < 2; f++) {
    bq[f].q[0] = *(const uint4*)&Qrow[f * 32 + g * 8];
    bq[f].q[1] = *(const uint4*)&Qrow[f * 32 + g * 8 + 16];
  }

  float mrun = -1e30f, lrun = 0.0f;
  v8f acc[4];
#pragma unroll
  for (int dt = 0; dt < 4; dt++)
#pragma unroll
    for (int e = 0; e < 8; e++) acc[dt][e] = 0.0f;

  const int nch = qblk / 32 + 2;  // chunks needed by the highest wave in block
  for (int c = 0; c < nch; ++c) {
    const int k0 = c * 32;
    // stage K (32 keys x 64 hd) and V^T (64 d x 32 keys)
#pragma unroll
    for (int i = 0; i < 2; i++) {
      int idx = tid + i * 128;                 // 0..255
      int row = idx >> 3, cc = (idx & 7) * 8;  // key row / hd chunk
      *(uint4*)&sK[row * LDK + cc] =
          *(const uint4*)&Kh[((size_t)bh * Sn + k0 + row) * HDn + cc];
      union { uint4 v4; u16 h[8]; } uv;
      uv.v4 = *(const uint4*)&Vh[((size_t)bh * Sn + k0 + row) * HDn + cc];
#pragma unroll
      for (int j = 0; j < 8; j++) sVt[(cc + j) * LDV + row] = uv.h[j];
    }
    __syncthreads();

    if (k0 <= q0 + 15) {  // this wave still needs keys from this chunk
      v8f st[2];
#pragma unroll
      for (int t = 0; t < 2; t++) {
#pragma unroll
        for (int e = 0; e < 8; e++) st[t][e] = 0.0f;
#pragma unroll
        for (int f = 0; f < 2; f++) {
          FragU ak;
          int kr = t * 16 + lr;  // local key row (A-matrix row)
          ak.q[0] = *(const uint4*)&sK[kr * LDK + f * 32 + g * 8];
          ak.q[1] = *(const uint4*)&sK[kr * LDK + f * 32 + g * 8 + 16];
          st[t] = __builtin_amdgcn_wmma_f32_16x16x32_bf16(
              false, ak.v, false, bq[f].v, (short)0, st[t], false, false);
        }
        if (k0 + t * 16 + 15 > q0) {  // tile crosses the causal diagonal
#pragma unroll
          for (int r = 0; r < 8; r++) {
            int key = k0 + t * 16 + g * 8 + r;
            if (key > q) st[t][r] = -1e30f;
          }
        }
      }
      // online softmax: each query's 32 chunk-scores live in lanes l, l+16
      float cmax = -1e30f;
#pragma unroll
      for (int t = 0; t < 2; t++)
#pragma unroll
        for (int r = 0; r < 8; r++) cmax = fmaxf(cmax, st[t][r]);
      cmax = fmaxf(cmax, __shfl_xor(cmax, 16, 32));
      float mnew = fmaxf(mrun, cmax);
      float alpha = __expf(mrun - mnew);
      float psum = 0.0f;
      FragU ap;  // P as A-fragment: lane-local repack (e<8 <- tile0, e>=8 <- tile1)
#pragma unroll
      for (int t = 0; t < 2; t++)
#pragma unroll
        for (int r = 0; r < 8; r++) {
          float p = __expf(st[t][r] - mnew);
          psum += p;
          ap.u[t * 8 + r] = f2bf(p);
        }
      psum += __shfl_xor(psum, 16, 32);
      lrun = lrun * alpha + psum;
      mrun = mnew;
      // rescale ctx accumulators: row M=r+8g needs alpha of query (r+8g)
#pragma unroll
      for (int r = 0; r < 8; r++) {
        float ra = __shfl(alpha, r + g * 8, 32);
#pragma unroll
        for (int dt = 0; dt < 4; dt++) acc[dt][r] *= ra;
      }
      // ctx += P @ V : V^T rows give contiguous B-fragment loads
#pragma unroll
      for (int dt = 0; dt < 4; dt++) {
        FragU bv;
        int d = dt * 16 + lr;
        bv.q[0] = *(const uint4*)&sVt[d * LDV + g * 8];
        bv.q[1] = *(const uint4*)&sVt[d * LDV + g * 8 + 16];
        acc[dt] = __builtin_amdgcn_wmma_f32_16x16x32_bf16(
            false, ap.v, false, bv.v, (short)0, acc[dt], false, false);
      }
    }
    __syncthreads();
  }

  // normalize and write ctx (bf16, [B,S,H*HD] row-major = [B,S,D])
  const int b = bh >> 4, h = bh & 15;
#pragma unroll
  for (int r = 0; r < 8; r++) {
    float rinv = 1.0f / __shfl(lrun, r + g * 8, 32);
    int srow = q0 + g * 8 + r;
    u16* dst = Ctx + ((size_t)(b * Sn + srow)) * Dn + h * 64;
#pragma unroll
    for (int dt = 0; dt < 4; dt++) dst[dt * 16 + lr] = f2bf(acc[dt][r] * rinv);
  }
}

// ---------------------------------------------------------------------------
// Fused residual + LayerNorm over D=1024 (one row per block, 256 threads)
// ---------------------------------------------------------------------------
template <bool WB>
__global__ __launch_bounds__(256, 1)
void ln_kernel(const float* __restrict__ X0, const float* __restrict__ Y,
               const float* __restrict__ gam, const float* __restrict__ bet,
               float* __restrict__ outf, u16* __restrict__ outb) {
  __shared__ float red[256];
  const int row = blockIdx.x, tid = threadIdx.x;
  const float* xr = X0 + (size_t)row * Dn;
  const float* yr = Y + (size_t)row * Dn;
  float x[4];
  float s = 0.0f;
#pragma unroll
  for (int j = 0; j < 4; j++) {
    x[j] = xr[tid + j * 256] + yr[tid + j * 256];
    s += x[j];
  }
  red[tid] = s;
  __syncthreads();
  for (int off = 128; off > 0; off >>= 1) {
    if (tid < off) red[tid] += red[tid + off];
    __syncthreads();
  }
  float mean = red[0] * (1.0f / Dn);
  __syncthreads();
  float vs = 0.0f;
#pragma unroll
  for (int j = 0; j < 4; j++) {
    float d = x[j] - mean;
    vs += d * d;
  }
  red[tid] = vs;
  __syncthreads();
  for (int off = 128; off > 0; off >>= 1) {
    if (tid < off) red[tid] += red[tid + off];
    __syncthreads();
  }
  float rstd = rsqrtf(red[0] * (1.0f / Dn) + 1e-7f);
#pragma unroll
  for (int j = 0; j < 4; j++) {
    int i = tid + j * 256;
    float yv = (x[j] - mean) * rstd * gam[i] + bet[i];
    outf[(size_t)row * Dn + i] = yv;
    if (WB) outb[(size_t)row * Dn + i] = f2bf(yv);
  }
}

// ---------------------------------------------------------------------------
// Launcher
// ---------------------------------------------------------------------------
extern "C" void kernel_launch(void* const* d_in, const int* in_sizes, int n_in,
                              void* d_out, int out_size, void* d_ws,
                              size_t ws_size, hipStream_t stream) {
  (void)in_sizes; (void)n_in; (void)out_size; (void)ws_size;
  const float* X   = (const float*)d_in[0];
  const float* Wq  = (const float*)d_in[1];
  const float* bq  = (const float*)d_in[2];
  const float* Wk  = (const float*)d_in[3];
  const float* bk  = (const float*)d_in[4];
  const float* Wv  = (const float*)d_in[5];
  const float* bv  = (const float*)d_in[6];
  const float* Wo  = (const float*)d_in[7];
  const float* bo  = (const float*)d_in[8];
  const float* g1  = (const float*)d_in[9];
  const float* b1  = (const float*)d_in[10];
  const float* Wup = (const float*)d_in[11];
  const float* bup = (const float*)d_in[12];
  const float* Wdn = (const float*)d_in[13];
  const float* bdn = (const float*)d_in[14];
  const float* g2  = (const float*)d_in[15];
  const float* b2  = (const float*)d_in[16];
  float* out = (float*)d_out;

  // workspace layout (bytes); bf16 buffers first, fp32 after (~136 MB total)
  char* ws = (char*)d_ws;
  u16* Xb   = (u16*)(ws + 0);           // 4096x1024
  u16* Wqb  = (u16*)(ws + 8388608);     // 1024x1024
  u16* Wkb  = (u16*)(ws + 10485760);
  u16* Wvb  = (u16*)(ws + 12582912);
  u16* Wob  = (u16*)(ws + 14680064);
  u16* Wupb = (u16*)(ws + 16777216);    // 4096x1024
  u16* Wdnb = (u16*)(ws + 25165824);    // 1024x4096
  u16* Qb   = (u16*)(ws + 33554432);    // [B,H,S,HD]
  u16* Kb   = (u16*)(ws + 41943040);
  u16* Vb   = (u16*)(ws + 50331648);
  u16* Ctxb = (u16*)(ws + 58720256);    // [B,S,D]
  u16* X1b  = (u16*)(ws + 67108864);    // 4096x1024
  u16* Upb  = (u16*)(ws + 75497472);    // 4096x4096
  float* Yf  = (float*)(ws + 109051904); // 4096x1024 fp32 scratch
  float* X1f = (float*)(ws + 125829120); // 4096x1024 fp32

  // 1) fp32 -> bf16 conversions
  cvt_kernel<<<1024, 256, 0, stream>>>(X, Xb, Mn * Dn);
  cvt_kernel<<<512, 256, 0, stream>>>(Wq, Wqb, Dn * Dn);
  cvt_kernel<<<512, 256, 0, stream>>>(Wk, Wkb, Dn * Dn);
  cvt_kernel<<<512, 256, 0, stream>>>(Wv, Wvb, Dn * Dn);
  cvt_kernel<<<512, 256, 0, stream>>>(Wo, Wob, Dn * Dn);
  cvt_kernel<<<1024, 256, 0, stream>>>(Wup, Wupb, FFn * Dn);
  cvt_kernel<<<1024, 256, 0, stream>>>(Wdn, Wdnb, Dn * FFn);

  // 2) QKV projections into head layout (Q pre-scaled by 1/sqrt(64) = 0.125)
  gemm_kernel<Mn, Dn, Dn, MODE_HEAD><<<dim3(8, 32), 128, 0, stream>>>(
      Xb, Wqb, bq, nullptr, Qb, 0.125f);
  gemm_kernel<Mn, Dn, Dn, MODE_HEAD><<<dim3(8, 32), 128, 0, stream>>>(
      Xb, Wkb, bk, nullptr, Kb, 1.0f);
  gemm_kernel<Mn, Dn, Dn, MODE_HEAD><<<dim3(8, 32), 128, 0, stream>>>(
      Xb, Wvb, bv, nullptr, Vb, 1.0f);

  // 3) causal flash attention -> Ctx bf16 [B,S,D]
  attn_kernel<<<dim3(Sn / 64, Bn * Hn), 128, 0, stream>>>(Qb, Kb, Vb, Ctxb);

  // 4) output projection (fp32) ; 5) residual + LN1 (fp32 + bf16)
  gemm_kernel<Mn, Dn, Dn, MODE_F32><<<dim3(8, 32), 128, 0, stream>>>(
      Ctxb, Wob, bo, Yf, nullptr, 1.0f);
  ln_kernel<true><<<Mn, 256, 0, stream>>>(X, Yf, g1, b1, X1f, X1b);

  // 6) FFN up + exact GELU (bf16) ; 7) FFN down (fp32)
  gemm_kernel<Mn, FFn, Dn, MODE_GELU><<<dim3(32, 32), 128, 0, stream>>>(
      X1b, Wupb, bup, nullptr, Upb, 1.0f);
  gemm_kernel<Mn, Dn, FFn, MODE_F32><<<dim3(8, 32), 128, 0, stream>>>(
      Upb, Wdnb, bdn, Yf, nullptr, 1.0f);

  // 8) residual + LN2 -> d_out (fp32)
  ln_kernel<false><<<Mn, 256, 0, stream>>>(X1f, Yf, g2, b2, out, nullptr);
}
